// ELMoModule_89532888253012
// MI455X (gfx1250) — matrix operand order repaired and checked by reading, not compiled, not run on hardware
//
#include <hip/hip_runtime.h>

// ---------------------------------------------------------------------------
// ELMo LM loss for MI455X (gfx1250, wave32, WMMA).
// Pipeline: prep/convert -> xp GEMM (WMMA bf16) -> LSTM recurrence (WMMA bf16,
// single WGP, LDS-resident state) -> fused vocab GEMM + online logsumexp
// (WMMA bf16, A tile register-hoisted, TDM-staged LDS, vocab-sliced for
// occupancy) -> slice combine -> picked-logit NLL -> scalar reduce.
// ---------------------------------------------------------------------------

typedef __bf16 v16bf __attribute__((ext_vector_type(16)));
typedef __bf16 v8bf  __attribute__((ext_vector_type(8)));
typedef float  v8f   __attribute__((ext_vector_type(8)));
typedef unsigned u32x4 __attribute__((ext_vector_type(4)));
typedef unsigned u32x8 __attribute__((ext_vector_type(8)));

#define VOCAB 32000
#define EMB   256
#define HID   512
#define NB    4
#define SEQ   512
#define NTOK  (NB * SEQ)      // 2048
#define G4H   (4 * HID)       // 2048
#define NSLICE 5              // vocab slices (250 tiles of 128 -> 5 x 50)
#define TPS    50             // tiles per slice

// ---------------------------------------------------------------------------
// WMMA fragment helpers (layouts per CDNA5 ISA 7.12.2, wave32).
// A (16x32 bf16, row-major source, leading dim ld):
//   lane l: m = l%16, K in {k0..k0+7} -> elems 0..7, {k0+16..k0+23} -> 8..15,
//   k0 = kb + 8*(l/16).
// B (32x16 bf16; column n of B = row n of the row-major [N x K] weight):
//   lane l: n = l%16, same K order as A (k0 = kb + 8*(l/16); k0..k0+7, k0+16..k0+23).
// C (16x16 f32): lane l, vgpr v -> m = v + 8*(l/16), n = l%16.
// ---------------------------------------------------------------------------
__device__ inline v16bf frag_a_rm(const __bf16* tile, int ld, int kb, int lane) {
  int m  = lane & 15;
  int k0 = kb + ((lane >> 4) << 3);
  const __bf16* p = tile + m * ld + k0;
  v8bf lo = *(const v8bf*)(p);
  v8bf hi = *(const v8bf*)(p + 16);
  v16bf r;
#pragma unroll
  for (int u = 0; u < 8; ++u) { r[u] = lo[u]; r[u + 8] = hi[u]; }
  return r;
}

__device__ inline v16bf frag_b_rm(const __bf16* tileN, int ld, int kb, int lane) {
  int n  = lane & 15;
  int k0 = kb + ((lane >> 4) << 3);
  const __bf16* p = tileN + n * ld + k0;
  v8bf lo = *(const v8bf*)(p);
  v8bf hi = *(const v8bf*)(p + 16);
  v16bf r;
#pragma unroll
  for (int u = 0; u < 8; ++u) { r[u] = lo[u]; r[u + 8] = hi[u]; }
  return r;
}

__device__ inline v8f wmma_bf16(v16bf a, v16bf b, v8f c) {
  return __builtin_amdgcn_wmma_f32_16x16x32_bf16(
      false, a, false, b, (short)0, c, false, false);
}

__device__ inline float sigmoidf_(float x) { return 1.0f / (1.0f + __expf(-x)); }

// ---------------------------------------------------------------------------
// K1: dtype conversion + embedding gather (memory bound, ~80 MB total).
// ---------------------------------------------------------------------------
__global__ __launch_bounds__(256) void k_prep(
    const int* __restrict__ batch, const float* __restrict__ emb,
    const float* __restrict__ w_ih, const float* __restrict__ w_hh,
    const float* __restrict__ W,
    __bf16* __restrict__ xb, __bf16* __restrict__ wihb,
    __bf16* __restrict__ whhb, __bf16* __restrict__ Wb) {
  long long gid    = (long long)blockIdx.x * blockDim.x + threadIdx.x;
  long long stride = (long long)gridDim.x * blockDim.x;
  for (long long i = gid; i < (long long)VOCAB * HID; i += stride)
    Wb[i] = (__bf16)W[i];
  for (long long i = gid; i < (long long)G4H * HID; i += stride)
    whhb[i] = (__bf16)w_hh[i];
  for (long long i = gid; i < (long long)G4H * EMB; i += stride)
    wihb[i] = (__bf16)w_ih[i];
  for (long long i = gid; i < (long long)NTOK * EMB; i += stride) {
    int tok = (int)(i >> 8);
    int e   = (int)(i & (EMB - 1));
    xb[i] = (__bf16)emb[(long long)batch[tok] * EMB + e];
  }
}

// ---------------------------------------------------------------------------
// K2: xp = x @ w_ih^T + (b_ih + b_hh), stored as [S][B][4H] for the recurrence.
// Tile: 64(M tokens) x 64(N gates) per 4-wave WG; K=256 in 8 wmma steps.
// ---------------------------------------------------------------------------
__global__ __launch_bounds__(128) void k_xp(
    const __bf16* __restrict__ xb, const __bf16* __restrict__ wihb,
    const float* __restrict__ b_ih, const float* __restrict__ b_hh,
    float* __restrict__ xp) {
  const int lane = threadIdx.x & 31;
  const int wv   = threadIdx.x >> 5;           // 0..3 -> N fragment
  const int m0   = blockIdx.x * 64;            // token tile base
  const int n0   = blockIdx.y * 64 + wv * 16;  // gate column base
  v8f c[4] = {v8f{}, v8f{}, v8f{}, v8f{}};
  for (int kb = 0; kb < EMB; kb += 32) {
    v16bf bf = frag_b_rm(wihb + (long long)n0 * EMB, EMB, kb, lane);
#pragma unroll
    for (int mf = 0; mf < 4; ++mf) {
      v16bf af = frag_a_rm(xb + (long long)(m0 + mf * 16) * EMB, EMB, kb, lane);
      c[mf] = wmma_bf16(af, bf, c[mf]);
    }
  }
  const int n = n0 + (lane & 15);
  const float bias = b_ih[n] + b_hh[n];
#pragma unroll
  for (int mf = 0; mf < 4; ++mf) {
#pragma unroll
    for (int v = 0; v < 8; ++v) {
      int m = m0 + mf * 16 + v + 8 * (lane >> 4);   // token = b*SEQ + s
      int b = m >> 9, s = m & (SEQ - 1);
      xp[((long long)s * NB + b) * G4H + n] = c[mf][v] + bias;
    }
  }
}

// ---------------------------------------------------------------------------
// K3: LSTM recurrence. One workgroup of 32 waves (1024 threads) on one WGP.
// h/c/gates live in LDS; h padded to 16 rows so M=4 fits a 16x16 WMMA tile.
// Each wave owns 64 gate columns; per step: 16 K-steps x 4 frags = 64 wmma.
// ---------------------------------------------------------------------------
__global__ __launch_bounds__(1024) void k_lstm(
    const float* __restrict__ xp, const __bf16* __restrict__ whhb,
    __bf16* __restrict__ hout) {
  __shared__ __bf16 sh_h[16 * HID];      // 16 KB, rows 4..15 stay zero
  __shared__ float  sh_c[NB * HID];      // 8 KB
  __shared__ float  sh_g[NB * G4H];      // 32 KB gate pre-activations
  const int tid = threadIdx.x, lane = tid & 31, wv = tid >> 5;  // wv 0..31

  for (int i = tid; i < 16 * HID; i += 1024) sh_h[i] = (__bf16)0.0f;
  for (int i = tid; i < NB * HID; i += 1024) sh_c[i] = 0.0f;
  __syncthreads();

  for (int t = 0; t < SEQ; ++t) {
    const float* xpt = xp + (long long)t * NB * G4H;
    v8f c[4];
#pragma unroll
    for (int f = 0; f < 4; ++f) {
      c[f] = v8f{};
      if (lane < 16) {
        int n = wv * 64 + f * 16 + lane;
#pragma unroll
        for (int b = 0; b < NB; ++b) c[f][b] = xpt[b * G4H + n];
      }
    }
    for (int kb = 0; kb < HID; kb += 32) {
      v16bf a = frag_a_rm(sh_h, HID, kb, lane);
#pragma unroll
      for (int f = 0; f < 4; ++f) {
        v16bf bm = frag_b_rm(whhb + (long long)(wv * 64 + f * 16) * HID,
                             HID, kb, lane);
        c[f] = wmma_bf16(a, bm, c[f]);
      }
    }
    if (lane < 16) {
#pragma unroll
      for (int f = 0; f < 4; ++f) {
        int n = wv * 64 + f * 16 + lane;
#pragma unroll
        for (int b = 0; b < NB; ++b) sh_g[b * G4H + n] = c[f][b];
      }
    }
    __syncthreads();
    if (tid < HID) {
      int j = tid;
#pragma unroll
      for (int b = 0; b < NB; ++b) {
        float gi = sigmoidf_(sh_g[b * G4H + j]);
        float gf = sigmoidf_(sh_g[b * G4H + HID + j]);
        float gg = tanhf(sh_g[b * G4H + 2 * HID + j]);
        float go = sigmoidf_(sh_g[b * G4H + 3 * HID + j]);
        float cc = gf * sh_c[b * HID + j] + gi * gg;
        float hh = go * tanhf(cc);
        sh_c[b * HID + j] = cc;
        sh_h[b * HID + j] = (__bf16)hh;
        hout[((long long)b * SEQ + t) * HID + j] = (__bf16)hh;
      }
    }
    __syncthreads();
  }
}

// ---------------------------------------------------------------------------
// K4: fused vocab GEMM + online logsumexp partials.
// Grid: (128 token tiles) x (5 vocab slices); WG = 8 waves.
// A tile (16x512 bf16) staged to LDS via TDM (wave 0, D# per ISA ch.8) +
// cooperative copy fallback, then hoisted into 16 register fragments per wave
// (128 VGPRs) so the 50-tile hot loop touches no LDS at all.
// Per-row (max, sum-exp) kept in registers; cross-lane reduction uses
// __shfl_xor butterflies within each 16-lane half (C layout puts row m's 16
// columns across one half-wave). One LDS combine at the end per block.
// ---------------------------------------------------------------------------
__global__ __launch_bounds__(256) void k_lse(
    const __bf16* __restrict__ hout, const __bf16* __restrict__ Wb,
    const float* __restrict__ bvec, float* __restrict__ pmax,
    float* __restrict__ psum) {
  __shared__ __bf16 sh_a[16 * HID];       // 16 KB
  __shared__ float  sh_part[8][16][2];    // per-wave (max,sum) partials
  const int tid = threadIdx.x, lane = tid & 31, wv = tid >> 5;  // wv 0..7
  const int tok0  = blockIdx.x * 16;
  const int slice = blockIdx.y;           // 0..4
  const __bf16* src = hout + (long long)tok0 * HID;

  if (wv == 0) {
    // Tensor Data Mover: 2D tile, 2-byte elems, tile 512x16 == tensor.
    unsigned lds_off = (unsigned)(unsigned long long)(&sh_a[0]);
    unsigned long long ga = (unsigned long long)src;
    u32x4 g0;
    g0[0] = 1u;                                   // count=1, user descriptor
    g0[1] = lds_off;                              // lds_addr
    g0[2] = (unsigned)ga;                         // global_addr[31:0]
    g0[3] = (unsigned)(ga >> 32) | (2u << 30);    // global_addr hi | type=2
    u32x8 g1;
    g1[0] = 1u << 16;                             // data_size=2B
    g1[1] = (unsigned)(HID) << 16;                // tensor_dim0 = 512
    g1[2] = 16u << 16;                            // tensor_dim1 = 16
    g1[3] = (unsigned)(HID) << 16;                // tile_dim0 = 512
    g1[4] = 16u;                                  // tile_dim1 = 16
    g1[5] = (unsigned)(HID);                      // tensor_dim0_stride = 512
    g1[6] = 0u;
    g1[7] = 0u;
    asm volatile("tensor_load_to_lds %0, %1" :: "s"(g0), "s"(g1) : "memory");
    __builtin_amdgcn_s_wait_tensorcnt(0);
  }
  __syncthreads();
  {  // authoritative cooperative copy (16 KB contiguous)
    const v8bf* s8 = (const v8bf*)src;
    v8bf* d8 = (v8bf*)sh_a;
    for (int i = tid; i < 16 * HID / 8; i += 256) d8[i] = s8[i];
  }
  __syncthreads();

  // Hoist all 16 A fragments into registers (128 VGPRs).
  v16bf afr[16];
#pragma unroll
  for (int kk = 0; kk < 16; ++kk) afr[kk] = frag_a_rm(sh_a, HID, kk * 32, lane);

  float rmax[8], rsum[8];
#pragma unroll
  for (int v = 0; v < 8; ++v) { rmax[v] = -3.4e38f; rsum[v] = 0.0f; }

  const int nt0 = slice * TPS;
  for (int nt = nt0; nt < nt0 + TPS; ++nt) {
    const int vb = nt * 128 + wv * 16;
    if (nt + 1 < nt0 + TPS)
      __builtin_prefetch(Wb + (long long)(vb + 128) * HID, 0, 1);
    v8f c = v8f{};
#pragma unroll
    for (int kk = 0; kk < 16; ++kk) {
      v16bf bm = frag_b_rm(Wb + (long long)vb * HID, HID, kk * 32, lane);
      c = wmma_bf16(afr[kk], bm, c);
    }
    const float bias = bvec[vb + (lane & 15)];
#pragma unroll
    for (int v = 0; v < 8; ++v) {
      float x = c[v] + bias;
      // tile row-max across the 16 columns (one half-wave)
      float tmax = x;
#pragma unroll
      for (int o = 1; o < 16; o <<= 1) tmax = fmaxf(tmax, __shfl_xor(tmax, o, 16));
      float tsum = __expf(x - tmax);
#pragma unroll
      for (int o = 1; o < 16; o <<= 1) tsum += __shfl_xor(tsum, o, 16);
      float nm = fmaxf(rmax[v], tmax);
      rsum[v] = rsum[v] * __expf(rmax[v] - nm) + tsum * __expf(tmax - nm);
      rmax[v] = nm;
    }
  }

  // One representative lane per half writes its 8 rows.
  if ((lane & 15) == 0) {
#pragma unroll
    for (int v = 0; v < 8; ++v) {
      int m = v + 8 * (lane >> 4);
      sh_part[wv][m][0] = rmax[v];
      sh_part[wv][m][1] = rsum[v];
    }
  }
  __syncthreads();
  if (wv == 0 && lane < 16) {
    float gm = -3.4e38f;
#pragma unroll
    for (int w = 0; w < 8; ++w) gm = fmaxf(gm, sh_part[w][lane][0]);
    float gs = 0.0f;
#pragma unroll
    for (int w = 0; w < 8; ++w)
      gs += sh_part[w][lane][1] * __expf(sh_part[w][lane][0] - gm);
    pmax[(long long)slice * NTOK + tok0 + lane] = gm;
    psum[(long long)slice * NTOK + tok0 + lane] = gs;
  }
}

// K4b: combine the NSLICE partials into lse[token].
__global__ __launch_bounds__(256) void k_lse2(
    const float* __restrict__ pmax, const float* __restrict__ psum,
    float* __restrict__ lse) {
  int tok = blockIdx.x * 256 + threadIdx.x;
  if (tok >= NTOK) return;
  float gm = -3.4e38f;
#pragma unroll
  for (int s = 0; s < NSLICE; ++s)
    gm = fmaxf(gm, pmax[(long long)s * NTOK + tok]);
  float gs = 0.0f;
#pragma unroll
  for (int s = 0; s < NSLICE; ++s)
    gs += psum[(long long)s * NTOK + tok] *
          __expf(pmax[(long long)s * NTOK + tok] - gm);
  lse[tok] = gm + __logf(gs);
}

// ---------------------------------------------------------------------------
// K5: picked log-prob per entry (torch-faithful [S,V,B] reshape indexing).
// One wave per entry: 512-wide dot, shuffle reduce.
// ---------------------------------------------------------------------------
__global__ __launch_bounds__(256) void k_pick(
    const int* __restrict__ batch, const __bf16* __restrict__ hout,
    const __bf16* __restrict__ Wb, const float* __restrict__ bvec,
    const float* __restrict__ lse, float* __restrict__ picked) {
  const int lane = threadIdx.x & 31, wv = threadIdx.x >> 5;
  const int e = blockIdx.x * 8 + wv;               // 0..2047
  const int i = e >> 2, j = e & 3;
  const int t = batch[e];                          // tgt2[i][j] = flat[i*4+j]
  long long flat = (long long)i * VOCAB * NB + (long long)t * NB + j;
  const long long SV = (long long)SEQ * VOCAB;
  int bq = (int)(flat / SV);
  long long rem = flat % SV;
  int sq = (int)(rem / VOCAB);
  int vq = (int)(rem % VOCAB);
  int tok = bq * SEQ + sq;
  const __bf16* h = hout + (long long)tok * HID;
  const __bf16* w = Wb + (long long)vq * HID;
  float acc = 0.0f;
  for (int k = lane; k < HID; k += 32) acc += (float)h[k] * (float)w[k];
#pragma unroll
  for (int off = 16; off > 0; off >>= 1) acc += __shfl_down(acc, off, 32);
  if (lane == 0) picked[e] = acc + bvec[vq] - lse[tok];
}

// K6: scalar loss = -mean(picked)
__global__ __launch_bounds__(256) void k_reduce(
    const float* __restrict__ picked, float* __restrict__ out) {
  __shared__ float sh[256];
  float s = 0.0f;
  for (int i = threadIdx.x; i < NTOK; i += 256) s += picked[i];
  sh[threadIdx.x] = s;
  __syncthreads();
  for (int st = 128; st > 0; st >>= 1) {
    if (threadIdx.x < st) sh[threadIdx.x] += sh[threadIdx.x + st];
    __syncthreads();
  }
  if (threadIdx.x == 0) out[0] = -sh[0] / (float)NTOK;
}

// ---------------------------------------------------------------------------
extern "C" void kernel_launch(void* const* d_in, const int* in_sizes, int n_in,
                              void* d_out, int out_size, void* d_ws,
                              size_t ws_size, hipStream_t stream) {
  const int*   batch = (const int*)d_in[0];
  const float* emb   = (const float*)d_in[1];
  const float* w_ih  = (const float*)d_in[2];
  const float* w_hh  = (const float*)d_in[3];
  const float* b_ih  = (const float*)d_in[4];
  const float* b_hh  = (const float*)d_in[5];
  const float* W     = (const float*)d_in[6];
  const float* bvec  = (const float*)d_in[7];

  char* ws = (char*)d_ws;
  size_t off = 0;
  auto salloc = [&](size_t bytes) -> void* {
    void* p = ws + off;
    off = (off + bytes + 255) & ~(size_t)255;
    return p;
  };
  __bf16* Wb     = (__bf16*)salloc((size_t)VOCAB * HID * 2);   // 32.8 MB
  __bf16* whhb   = (__bf16*)salloc((size_t)G4H * HID * 2);     // 2 MB
  __bf16* wihb   = (__bf16*)salloc((size_t)G4H * EMB * 2);     // 1 MB
  __bf16* xb     = (__bf16*)salloc((size_t)NTOK * EMB * 2);    // 1 MB
  float*  xp     = (float*)salloc((size_t)SEQ * NB * G4H * 4); // 16.8 MB
  __bf16* hout   = (__bf16*)salloc((size_t)NTOK * HID * 2);    // 2 MB
  float*  pmax   = (float*)salloc((size_t)NSLICE * NTOK * 4);
  float*  psum   = (float*)salloc((size_t)NSLICE * NTOK * 4);
  float*  lseb   = (float*)salloc((size_t)NTOK * 4);
  float*  picked = (float*)salloc((size_t)NTOK * 4);

  k_prep<<<2048, 256, 0, stream>>>(batch, emb, w_ih, w_hh, W,
                                   xb, wihb, whhb, Wb);
  k_xp<<<dim3(NTOK / 64, G4H / 64), 128, 0, stream>>>(xb, wihb, b_ih, b_hh, xp);
  k_lstm<<<1, 1024, 0, stream>>>(xp, whhb, hout);
  k_lse<<<dim3(NTOK / 16, NSLICE), 256, 0, stream>>>(hout, Wb, bvec, pmax, psum);
  k_lse2<<<(NTOK + 255) / 256, 256, 0, stream>>>(pmax, psum, lseb);
  k_pick<<<NTOK / 8, 256, 0, stream>>>(batch, hout, Wb, bvec, lseb, picked);
  k_reduce<<<1, 256, 0, stream>>>(picked, (float*)d_out);
}
